// ReservoirNetwork_74981539053641
// MI455X (gfx1250) — compile-verified, matrix operand order checked by reading, not run on hardware
//
#include <hip/hip_runtime.h>

typedef __attribute__((ext_vector_type(16))) __bf16        v16bf;
typedef __attribute__((ext_vector_type(8)))  float         v8f;
typedef __attribute__((ext_vector_type(4)))  unsigned int  v4u;

#define R_DIM   1024
#define IN_DIM  128
#define BATCHN  256
#define SEQ     512
#define KTILES  36                 // (1024 + 128) / 32
#define NTILES  64                 // 1024 / 16
#define FRAG_BYTES 1024            // per 32x16 B block: 32 lanes * 32B
#define KSTEP_BYTES ((size_t)NTILES * FRAG_BYTES)   // 64KB per K-chunk of frags

__device__ __forceinline__ unsigned short f32_to_bf16_rne(float f) {
  unsigned int u = __builtin_bit_cast(unsigned int, f);
  u += 0x7FFFu + ((u >> 16) & 1u);
  return (unsigned short)(u >> 16);
}
__device__ __forceinline__ float bf16hi_to_f32(unsigned short h) {
  return __builtin_bit_cast(float, (unsigned int)h << 16);
}

union BfFrag { v16bf v; unsigned short s[16]; v4u q[2]; };

// --------------------------------------------------------------------------
// One-time prep: hi/lo bf16 B-fragments of W_big = [W_res ; W_in^T] (1152x1024)
// pre-shuffled into the WMMA 32x16 B lane layout: lane L holds column N=L%16,
// element e -> K = (e<8?0:16) + (L>=16?8:0) + (e&7). 1KB per (ktile,ntile)
// block, 32B contiguous per lane -> pure b128 traffic in the hot loop.
// --------------------------------------------------------------------------
__global__ void build_w_frags(const float* __restrict__ Wres,
                              const float* __restrict__ Win,
                              unsigned short* __restrict__ whi,
                              unsigned short* __restrict__ wlo) {
  int wid  = blockIdx.x * (blockDim.x >> 5) + (threadIdx.x >> 5); // 0..2303
  int lane = threadIdx.x & 31;
  int ktile = wid >> 6;
  int ntile = wid & 63;
  int n     = ntile * 16 + (lane & 15);
  int hi8   = (lane & 16) >> 1;

  BfFrag fh, fl;
#pragma unroll
  for (int e = 0; e < 16; ++e) {
    int kk = ktile * 32 + ((e & 8) << 1) + hi8 + (e & 7);
    float w = (kk < R_DIM) ? Wres[(size_t)kk * R_DIM + n]
                           : Win[(size_t)n * IN_DIM + (kk - R_DIM)];
    unsigned short h = f32_to_bf16_rne(w);
    fh.s[e] = h;
    fl.s[e] = f32_to_bf16_rne(w - bf16hi_to_f32(h));
  }
  size_t off = (size_t)wid * FRAG_BYTES + (size_t)lane * 32;
  *(v16bf*)((char*)whi + off) = fh.v;
  *(v16bf*)((char*)wlo + off) = fl.v;
}

// One-time prep: x[b][i][t] fp32 -> xhi/xlo[t][b][i] bf16 (write-coalesced;
// strided reads are absorbed by the 192MB L2). Makes the per-step input tail
// pure aligned b128 loads with zero conversion VALU.
__global__ void convert_x(const float* __restrict__ x,
                          unsigned short* __restrict__ xhi,
                          unsigned short* __restrict__ xlo) {
  size_t gid = (size_t)blockIdx.x * blockDim.x + threadIdx.x; // over T*B*I
  int i = (int)(gid & (IN_DIM - 1));
  int b = (int)((gid >> 7) & (BATCHN - 1));
  int t = (int)(gid >> 15);
  float v = x[((size_t)b * IN_DIM + i) * SEQ + t];
  unsigned short h = f32_to_bf16_rne(v);
  xhi[gid] = h;
  xlo[gid] = f32_to_bf16_rne(v - bf16hi_to_f32(h));
}

__global__ void zero_u32(unsigned int* __restrict__ p) {
  p[(size_t)blockIdx.x * blockDim.x + threadIdx.x] = 0u;
}

// ---- hot-loop helpers ----------------------------------------------------
__device__ __forceinline__ void load_chunk(const char* ahp, const char* alp, int akt,
                                           const char* whi_b, const char* wlo_b,
                                           size_t fragbase, int kt,
                                           BfFrag& ahi, BfFrag& alo, v16bf (&b)[4]) {
  size_t ko = (size_t)akt * 64;                 // 32 bf16 per chunk
  ahi.q[0] = *(const v4u*)(ahp + ko);
  ahi.q[1] = *(const v4u*)(ahp + ko + 32);
  alo.q[0] = *(const v4u*)(alp + ko);
  alo.q[1] = *(const v4u*)(alp + ko + 32);
  size_t base = fragbase + (size_t)kt * KSTEP_BYTES;
  b[0] = *(const v16bf*)(whi_b + base);
  b[1] = *(const v16bf*)(whi_b + base + FRAG_BYTES);
  b[2] = *(const v16bf*)(wlo_b + base);
  b[3] = *(const v16bf*)(wlo_b + base + FRAG_BYTES);
}

__device__ __forceinline__ void mma_chunk(const BfFrag& ahi, const BfFrag& alo,
                                          const v16bf (&b)[4], v8f (&acc)[2]) {
  // split precision: hi*Whi + lo*Whi + hi*Wlo, f32 accumulate; interleave the
  // two accumulators to avoid back-to-back RAW on the same D registers.
  acc[0] = __builtin_amdgcn_wmma_f32_16x16x32_bf16(false, ahi.v, false, b[0], (short)0, acc[0], false, false);
  acc[1] = __builtin_amdgcn_wmma_f32_16x16x32_bf16(false, ahi.v, false, b[1], (short)0, acc[1], false, false);
  acc[0] = __builtin_amdgcn_wmma_f32_16x16x32_bf16(false, alo.v, false, b[0], (short)0, acc[0], false, false);
  acc[1] = __builtin_amdgcn_wmma_f32_16x16x32_bf16(false, alo.v, false, b[1], (short)0, acc[1], false, false);
  acc[0] = __builtin_amdgcn_wmma_f32_16x16x32_bf16(false, ahi.v, false, b[2], (short)0, acc[0], false, false);
  acc[1] = __builtin_amdgcn_wmma_f32_16x16x32_bf16(false, ahi.v, false, b[3], (short)0, acc[1], false, false);
}

// --------------------------------------------------------------------------
// One step: out = tanh( [H | U] @ [W_res ; W_in^T] ). H kept as bf16 hi/lo
// arrays (row-major [256][1024]) -> A fragments are raw b128 loads. K-loop is
// double-buffered: chunk kt+1 loads in flight under chunk kt's 6 WMMAs.
// Grid (16,8), block 128 (4 waves); wave owns a 16(M) x 32(N) strip.
// Uhi/Ulo non-null: preconverted bf16 input tail. Fout non-null: fp32 store.
// --------------------------------------------------------------------------
__global__ __launch_bounds__(128)
void reservoir_step(const unsigned short* __restrict__ Hhi,
                    const unsigned short* __restrict__ Hlo,
                    const unsigned short* __restrict__ Uhi,
                    const unsigned short* __restrict__ Ulo,
                    const float* __restrict__ U, int u_row_stride, int u_col_stride,
                    const unsigned short* __restrict__ whi,
                    const unsigned short* __restrict__ wlo,
                    unsigned short* __restrict__ HhiOut,
                    unsigned short* __restrict__ HloOut,
                    float* __restrict__ Fout) {
  int lane   = threadIdx.x & 31;
  int wave   = threadIdx.x >> 5;
  int hi8    = (lane & 16) >> 1;            // 8 for lanes 16..31
  int mbase  = blockIdx.x * 16;
  int m      = mbase + (lane & 15);
  int ntile0 = blockIdx.y * 8 + wave * 2;

  v8f acc[2] = {};
  const char* whi_b = (const char*)whi;
  const char* wlo_b = (const char*)wlo;
  const char* ahp = (const char*)Hhi + ((size_t)m * R_DIM + hi8) * 2;
  const char* alp = (const char*)Hlo + ((size_t)m * R_DIM + hi8) * 2;
  size_t fragbase = (size_t)ntile0 * FRAG_BYTES + (size_t)lane * 32;

  BfFrag a0h, a0l, a1h, a1l;
  v16bf b0[4], b1[4];

  // ---- main K loop over H (32 chunks), double-buffered ----
  load_chunk(ahp, alp, 0, whi_b, wlo_b, fragbase, 0, a0h, a0l, b0);
  for (int kt = 0; kt < 32; kt += 2) {
    load_chunk(ahp, alp, kt + 1, whi_b, wlo_b, fragbase, kt + 1, a1h, a1l, b1);
    mma_chunk(a0h, a0l, b0, acc);
    if (kt + 2 < 32)
      load_chunk(ahp, alp, kt + 2, whi_b, wlo_b, fragbase, kt + 2, a0h, a0l, b0);
    mma_chunk(a1h, a1l, b1, acc);
  }

  // ---- input-projection tail (4 chunks, K = 1024..1151) ----
  if (Uhi != nullptr) {
    const char* uhp = (const char*)Uhi + ((size_t)m * IN_DIM + hi8) * 2;
    const char* ulp = (const char*)Ulo + ((size_t)m * IN_DIM + hi8) * 2;
    load_chunk(uhp, ulp, 0, whi_b, wlo_b, fragbase, 32, a0h, a0l, b0);
#pragma unroll
    for (int kt = 32; kt < 36; ++kt) {
      if (kt + 1 < 36)
        load_chunk(uhp, ulp, kt - 31, whi_b, wlo_b, fragbase, kt + 1, a1h, a1l, b1);
      mma_chunk(a0h, a0l, b0, acc);
      a0h = a1h; a0l = a1l;
      b0[0] = b1[0]; b0[1] = b1[1]; b0[2] = b1[2]; b0[3] = b1[3];
    }
  } else {
    // generic strided fp32 gather (cls/dist finals, or small-ws fallback)
    for (int kt = 32; kt < 36; ++kt) {
      int i0 = (kt - 32) * 32 + hi8;
      const float* up = U + (size_t)m * u_row_stride;
#pragma unroll
      for (int j = 0; j < 8; ++j) {
        float v0 = up[(size_t)(i0 + j)      * u_col_stride];
        float v1 = up[(size_t)(i0 + 16 + j) * u_col_stride];
        unsigned short h0 = f32_to_bf16_rne(v0);
        unsigned short h1 = f32_to_bf16_rne(v1);
        a0h.s[j]     = h0;  a0l.s[j]     = f32_to_bf16_rne(v0 - bf16hi_to_f32(h0));
        a0h.s[8 + j] = h1;  a0l.s[8 + j] = f32_to_bf16_rne(v1 - bf16hi_to_f32(h1));
      }
      size_t base = fragbase + (size_t)kt * KSTEP_BYTES;
      b0[0] = *(const v16bf*)(whi_b + base);
      b0[1] = *(const v16bf*)(whi_b + base + FRAG_BYTES);
      b0[2] = *(const v16bf*)(wlo_b + base);
      b0[3] = *(const v16bf*)(wlo_b + base + FRAG_BYTES);
      mma_chunk(a0h, a0l, b0, acc);
    }
  }

  // ---- epilogue: tanh, then fp32 store (finals) or bf16 hi/lo split store ----
  if (Fout != nullptr) {
#pragma unroll
    for (int nt = 0; nt < 2; ++nt) {
      int ncol = (ntile0 + nt) * 16 + (lane & 15);
#pragma unroll
      for (int v = 0; v < 8; ++v)
        Fout[(size_t)(mbase + v + hi8) * R_DIM + ncol] = tanhf(acc[nt][v]);
    }
  } else {
#pragma unroll
    for (int nt = 0; nt < 2; ++nt) {
      int ncol = (ntile0 + nt) * 16 + (lane & 15);
#pragma unroll
      for (int v = 0; v < 8; ++v) {
        float t = tanhf(acc[nt][v]);
        unsigned short h = f32_to_bf16_rne(t);
        unsigned short l = f32_to_bf16_rne(t - bf16hi_to_f32(h));
        size_t idx = (size_t)(mbase + v + hi8) * R_DIM + ncol;
        HhiOut[idx] = h;
        HloOut[idx] = l;
      }
    }
  }
}

// --------------------------------------------------------------------------
extern "C" void kernel_launch(void* const* d_in, const int* in_sizes, int n_in,
                              void* d_out, int out_size, void* d_ws, size_t ws_size,
                              hipStream_t stream) {
  const float* x    = (const float*)d_in[0];   // [256,128,512]
  const float* cls  = (const float*)d_in[1];   // [256,128]
  const float* dist = (const float*)d_in[2];   // [256,128]
  const float* Wres = (const float*)d_in[3];   // [1024,1024]
  const float* Win  = (const float*)d_in[4];   // [1024,128]
  float* out = (float*)d_out;                  // [2,256,1024] fp32

  char* ws = (char*)d_ws;
  const size_t frag_bytes = (size_t)KTILES * NTILES * FRAG_BYTES;   // 2.25 MB each
  const size_t hElems = (size_t)BATCHN * R_DIM;                     // 262144
  const size_t xElems = (size_t)SEQ * BATCHN * IN_DIM;              // 16,777,216
  unsigned short* whi  = (unsigned short*)ws;
  unsigned short* wlo  = (unsigned short*)(ws + frag_bytes);
  unsigned short* hhi0 = (unsigned short*)(ws + 2 * frag_bytes);
  unsigned short* hlo0 = hhi0 + hElems;
  unsigned short* hhi1 = hlo0 + hElems;
  unsigned short* hlo1 = hhi1 + hElems;
  unsigned short* xhi  = hlo1 + hElems;
  unsigned short* xlo  = xhi + xElems;
  const size_t need_fast = 2 * frag_bytes + (4 * hElems + 2 * xElems) * sizeof(unsigned short);
  const bool fast = (ws_size >= need_fast);

  build_w_frags<<<dim3(288), dim3(256), 0, stream>>>(Wres, Win, whi, wlo);
  zero_u32<<<dim3(512), dim3(512), 0, stream>>>((unsigned int*)hhi0);  // hhi0+hlo0
  if (fast)
    convert_x<<<dim3((unsigned)(xElems / 256)), dim3(256), 0, stream>>>(x, xhi, xlo);

  dim3 grid(BATCHN / 16, R_DIM / 128);  // 16 x 8 workgroups, 512 waves
  dim3 block(128);                      // 4 waves

  const unsigned short* hi_in = hhi0; const unsigned short* lo_in = hlo0;
  unsigned short* hi_out = hhi1;       unsigned short* lo_out = hlo1;
  for (int t = 0; t < SEQ; ++t) {
    const unsigned short* uhi = fast ? xhi + (size_t)t * BATCHN * IN_DIM : nullptr;
    const unsigned short* ulo = fast ? xlo + (size_t)t * BATCHN * IN_DIM : nullptr;
    reservoir_step<<<grid, block, 0, stream>>>(hi_in, lo_in, uhi, ulo,
                                               x + t, IN_DIM * SEQ, SEQ,
                                               whi, wlo, hi_out, lo_out, nullptr);
    const unsigned short* th = hi_in; const unsigned short* tl = lo_in;
    hi_in = hi_out; lo_in = lo_out;
    hi_out = (unsigned short*)th; lo_out = (unsigned short*)tl;
  }
  // final projections -> fp32 into d_out (same GEMM, U = cls / dist)
  reservoir_step<<<grid, block, 0, stream>>>(hi_in, lo_in, nullptr, nullptr,
                                             cls, IN_DIM, 1,
                                             whi, wlo, nullptr, nullptr, out);
  reservoir_step<<<grid, block, 0, stream>>>(hi_in, lo_in, nullptr, nullptr,
                                             dist, IN_DIM, 1,
                                             whi, wlo, nullptr, nullptr, out + hElems);
}
